// Encoder_14096082665773
// MI455X (gfx1250) — matrix operand order repaired and checked by reading, not compiled, not run on hardware
//
#include <hip/hip_runtime.h>
#include <hip/hip_bf16.h>
#include <math.h>

typedef __attribute__((ext_vector_type(16))) _Float16 v16h;
typedef __attribute__((ext_vector_type(8)))  float    v8f;

#define BB 8           // batch
#define FPS_T 256

// ---------------------------------------------------------------------------
// (B,3,N) -> (B,N,3)
__global__ void k_transpose_in(const float* __restrict__ src, float* __restrict__ dst, int N) {
    long i = (long)blockIdx.x * blockDim.x + threadIdx.x;
    long total = (long)BB * N * 3;
    if (i >= total) return;
    int c = i % 3; long bn = i / 3; int n = bn % N; int b = bn / N;
    dst[i] = src[((long)b * 3 + c) * N + n];
}

// (B,S,C) -> (B,C,S)
__global__ void k_transpose_out(const float* __restrict__ src, float* __restrict__ dst, int S, int C) {
    long i = (long)blockIdx.x * blockDim.x + threadIdx.x;
    long total = (long)BB * S * C;
    if (i >= total) return;
    int s = i % S; long bc = i / S; int c = bc % C; int b = bc / C;
    dst[i] = src[((long)b * S + s) * C + c];
}

// ---------------------------------------------------------------------------
// Farthest point sampling: one block per batch, dist kept in LDS.
__global__ void k_fps(const float* __restrict__ pts, int N, int S, int* __restrict__ idx_out) {
    __shared__ float sdist[8192];
    __shared__ float smax[FPS_T];
    __shared__ int   simax[FPS_T];
    __shared__ int   sfar;
    const int b = blockIdx.x;
    const int t = threadIdx.x;
    const float* p = pts + (long)b * N * 3;
    for (int i = t; i < N; i += FPS_T) sdist[i] = 1e10f;
    if (t == 0) sfar = 0;
    __syncthreads();
    for (int it = 0; it < S; ++it) {
        int far = sfar;
        if (t == 0) idx_out[b * S + it] = far;
        float cx = p[far * 3 + 0], cy = p[far * 3 + 1], cz = p[far * 3 + 2];
        float lm = -1.0f; int li = 0;
        for (int i = t; i < N; i += FPS_T) {
            float dx = p[i * 3 + 0] - cx, dy = p[i * 3 + 1] - cy, dz = p[i * 3 + 2] - cz;
            float nd = fminf(sdist[i], dx * dx + dy * dy + dz * dz);
            sdist[i] = nd;
            if (nd > lm) { lm = nd; li = i; }
        }
        smax[t] = lm; simax[t] = li;
        __syncthreads();
        for (int s = FPS_T / 2; s > 0; s >>= 1) {
            if (t < s) {
                float om = smax[t + s]; int oi = simax[t + s];
                if (om > smax[t] || (om == smax[t] && oi < simax[t])) { smax[t] = om; simax[t] = oi; }
            }
            __syncthreads();
        }
        if (t == 0) sfar = simax[0];
        __syncthreads();
    }
}

__global__ void k_gather_newxyz(const float* __restrict__ pts, const int* __restrict__ fps,
                                int N, int S, float* __restrict__ nx) {
    int i = blockIdx.x * blockDim.x + threadIdx.x;
    if (i >= BB * S) return;
    int b = i / S;
    int g = fps[i];
    const float* src = pts + ((long)b * N + g) * 3;
    nx[i * 3 + 0] = src[0]; nx[i * 3 + 1] = src[1]; nx[i * 3 + 2] = src[2];
}

// ---------------------------------------------------------------------------
// Ball query: one wave32 per (b,s); first-K in-radius indices in index order.
__global__ void k_ball_query(const float* __restrict__ pts, const float* __restrict__ nx,
                             int N, int S, int K, float r2, int* __restrict__ gidx) {
    int ws = blockIdx.x * blockDim.y + threadIdx.y;   // b*S+s
    if (ws >= BB * S) return;
    int lane = threadIdx.x;
    int b = ws / S;
    const float* p = pts + (long)b * N * 3;
    float cx = nx[ws * 3 + 0], cy = nx[ws * 3 + 1], cz = nx[ws * 3 + 2];
    int* out = gidx + (long)ws * K;
    int count = 0, firstIdx = -1;
    for (int base = 0; base < N && count < K; base += 32) {
        int i = base + lane;
        bool in = false;
        if (i < N) {
            float dx = p[i * 3 + 0] - cx, dy = p[i * 3 + 1] - cy, dz = p[i * 3 + 2] - cz;
            in = (dx * dx + dy * dy + dz * dz) <= r2;
        }
        unsigned mask = __builtin_amdgcn_ballot_w32(in);
        if (firstIdx < 0 && mask) firstIdx = base + __ffs(mask) - 1;
        if (in) {
            int pos = count + __popc(mask & ((1u << lane) - 1u));
            if (pos < K) out[pos] = i;
        }
        count += __popc(mask);
    }
    int first = (firstIdx >= 0) ? firstIdx : (N - 1);
    for (int pos = count + lane; pos < K; pos += 32) out[pos] = first;
}

// ---------------------------------------------------------------------------
// Group: build f16 A matrix rows [feat(C) | rel_xyz(3) | zero pad] of width CinPad.
__global__ void k_gather_group(const float* __restrict__ feats, int C, int fstride,
                               const float* __restrict__ pts, const float* __restrict__ nx,
                               const int* __restrict__ gidx, int N, int S, int K, int CinPad,
                               _Float16* __restrict__ A) {
    long row = blockIdx.x;                 // (b*S+s)*K + k
    int ws = row / K;
    int b = ws / S;
    int g = gidx[row];
    const float* f = feats + ((long)b * N + g) * fstride;
    const float* pp = pts + ((long)b * N + g) * 3;
    const float* cc = nx + (long)ws * 3;
    _Float16* out = A + row * CinPad;
    for (int col = threadIdx.x; col < CinPad; col += blockDim.x) {
        float v;
        if (col < C)          v = f[col];
        else if (col < C + 3) v = pp[col - C] - cc[col - C];
        else                  v = 0.0f;
        out[col] = (_Float16)v;
    }
}

// ---------------------------------------------------------------------------
// Pack f32 weights (Cout,Cin) row-major into zero-padded f16 (CoutPad,CinPad).
__global__ void k_pack_w(const float* __restrict__ W, int Cin, int Cout,
                         int CinPad, int CoutPad, _Float16* __restrict__ Wp) {
    long i = (long)blockIdx.x * blockDim.x + threadIdx.x;
    long total = (long)CoutPad * CinPad;
    if (i >= total) return;
    int c = i % CinPad; int d = i / CinPad;
    float v = (c < Cin && d < Cout) ? W[(long)d * Cin + c] : 0.0f;
    Wp[i] = (_Float16)v;
}

// ---------------------------------------------------------------------------
// WMMA GEMM + bias + BN + ReLU: each wave owns 4 M-tiles x 2 N-tiles
// (8 WMMAs per k-step; A fragments reused 2x, B fragments reused 4x).
// A: (M, CinPad) f16 row-major.  Wp: (CoutPad, CinPad) f16 padded.
// Y[M,Cout] = relu(bn(A * Wp^T + b)); padded channels written as 0.
__global__ __launch_bounds__(128) void k_gemm_bn_relu(
        const _Float16* __restrict__ A, int lda,
        const _Float16* __restrict__ Wp,
        const float* __restrict__ bias,
        const float* __restrict__ gamma, const float* __restrict__ beta,
        const float* __restrict__ mean, const float* __restrict__ var,
        _Float16* __restrict__ Out, int ldo, int Cout) {
    const int lane = threadIdx.x;                    // 0..31
    const int wave = threadIdx.y;                    // 0..3
    const int row0 = (blockIdx.x * 4 + wave) * 64;   // 4 M-tiles per wave
    const int m    = lane & 15;
    const int half = lane >> 4;
    const int d0   = blockIdx.y * 32 + m;            // N-tile pair: d0, d0+16
    const int d1   = d0 + 16;

    const _Float16* wr0 = Wp + (long)d0 * lda + half * 16;
    const _Float16* wr1 = Wp + (long)d1 * lda + half * 16;

    v8f acc0[4] = {{}, {}, {}, {}};
    v8f acc1[4] = {{}, {}, {}, {}};
    const int ktiles = lda >> 5;                     // lda multiple of 32
    for (int kt = 0; kt < ktiles; ++kt) {
        v16h b0, b1;
        #pragma unroll
        for (int e = 0; e < 16; ++e) {               // K = 16*half + e  (same k order as A)
            b0[e] = wr0[kt * 32 + e];
            b1[e] = wr1[kt * 32 + e];
        }
        #pragma unroll
        for (int t = 0; t < 4; ++t) {
            const _Float16* arow = A + (long)(row0 + t * 16 + m) * lda + kt * 32;
            v16h a;
            #pragma unroll
            for (int e = 0; e < 16; ++e) {           // K = 16*half + e
                int kl = half * 16 + e;
                a[e] = arow[kl];
            }
            acc0[t] = __builtin_amdgcn_wmma_f32_16x16x32_f16(
                          false, a, false, b0, (short)0, acc0[t], false, false);
            acc1[t] = __builtin_amdgcn_wmma_f32_16x16x32_f16(
                          false, a, false, b1, (short)0, acc1[t], false, false);
        }
    }

    float g0 = 0.f, bt0 = 0.f, mn0 = 0.f, iv0 = 0.f, bs0 = 0.f;
    float g1 = 0.f, bt1 = 0.f, mn1 = 0.f, iv1 = 0.f, bs1 = 0.f;
    const bool v0 = d0 < Cout, v1 = d1 < Cout;
    if (v0) { g0 = gamma[d0]; bt0 = beta[d0]; mn0 = mean[d0];
              iv0 = rsqrtf(var[d0] + 1e-5f); bs0 = bias[d0]; }
    if (v1) { g1 = gamma[d1]; bt1 = beta[d1]; mn1 = mean[d1];
              iv1 = rsqrtf(var[d1] + 1e-5f); bs1 = bias[d1]; }

    #pragma unroll
    for (int t = 0; t < 4; ++t) {
        #pragma unroll
        for (int r = 0; r < 8; ++r) {
            int mr = half * 8 + r;                   // ISA f32 C/D layout
            long rowOff = (long)(row0 + t * 16 + mr) * ldo;
            float y0 = v0 ? fmaxf(g0 * (acc0[t][r] + bs0 - mn0) * iv0 + bt0, 0.0f) : 0.0f;
            float y1 = v1 ? fmaxf(g1 * (acc1[t][r] + bs1 - mn1) * iv1 + bt1, 0.0f) : 0.0f;
            Out[rowOff + d0] = (_Float16)y0;
            Out[rowOff + d1] = (_Float16)y1;
        }
    }
}

// ---------------------------------------------------------------------------
// Max over group dim K -> f32 feature buffer at channel offset.
__global__ void k_maxpool(const _Float16* __restrict__ act, int ldo, int Cout, int K,
                          float* __restrict__ fout, int Ctot, int choff) {
    long ws = blockIdx.x;                    // b*S+s
    for (int dch = threadIdx.x; dch < Cout; dch += blockDim.x) {
        float mx = -1e30f;
        for (int k = 0; k < K; ++k)
            mx = fmaxf(mx, (float)act[((long)ws * K + k) * ldo + dch]);
        fout[ws * (long)Ctot + choff + dch] = mx;
    }
}

// ---------------------------------------------------------------------------
static inline int pad32(int x) { return (x + 31) & ~31; }

extern "C" void kernel_launch(void* const* d_in, const int* in_sizes, int n_in,
                              void* d_out, int out_size, void* d_ws, size_t ws_size,
                              hipStream_t stream) {
    (void)in_sizes; (void)n_in; (void)out_size; (void)ws_size;
    const float* xyz_in = (const float*)d_in[0];
    // params: 1 + ((li*2+br)*3+ly)*6 + {W,b,gamma,beta,mean,var}
    auto P = [&](int li, int br, int ly, int which) -> const float* {
        return (const float*)d_in[1 + ((li * 2 + br) * 3 + ly) * 6 + which];
    };

    static const int   S_arr[3]  = {1024, 512, 256};
    static const int   N_arr[3]  = {8192, 1024, 512};
    static const float rad[3][2] = {{0.025f, 0.05f}, {0.05f, 0.1f}, {0.1f, 0.2f}};
    static const int   Karr[2]   = {16, 32};
    static const int   C0[3]     = {3, 64, 256};
    static const int   MLP[3][3] = {{3, 32, 32}, {64, 64, 128}, {256, 196, 256}};

    // workspace layout
    char* ws = (char*)d_ws;
    size_t off = 0;
    auto alloc = [&](size_t bytes) { void* p = ws + off; off += (bytes + 255) & ~(size_t)255; return p; };
    float* pts1    = (float*)alloc((size_t)BB * 8192 * 3 * 4);
    float* nxyz[3];
    nxyz[0] = (float*)alloc((size_t)BB * 1024 * 3 * 4);
    nxyz[1] = (float*)alloc((size_t)BB * 512 * 3 * 4);
    nxyz[2] = (float*)alloc((size_t)BB * 256 * 3 * 4);
    float* feats2  = (float*)alloc((size_t)BB * 1024 * 64 * 4);
    float* feats3  = (float*)alloc((size_t)BB * 512 * 256 * 4);
    float* feats4  = (float*)alloc((size_t)BB * 256 * 512 * 4);
    int*   fpsidx  = (int*)alloc((size_t)BB * 1024 * 4);
    int*   gidx    = (int*)alloc((size_t)BB * 1024 * 32 * 4);
    _Float16* Wp   = (_Float16*)alloc(256 * 1024);          // packed weights (max 147KB)
    _Float16* bufA = (_Float16*)alloc(40u * 1024 * 1024);
    _Float16* bufB = (_Float16*)alloc(40u * 1024 * 1024);

    // transpose input xyz (B,3,N)->(B,N,3)
    {
        long total = (long)BB * 8192 * 3;
        k_transpose_in<<<(total + 255) / 256, 256, 0, stream>>>(xyz_in, pts1, 8192);
    }

    const float* xyzL[3]   = {pts1, nxyz[0], nxyz[1]};
    const float* featsL[3] = {pts1, feats2, feats3};
    const int    fstride[3] = {3, 64, 256};
    float* featsOut[3] = {feats2, feats3, feats4};
    const int   CtotOut[3] = {64, 256, 512};

    for (int li = 0; li < 3; ++li) {
        const int S = S_arr[li], N = N_arr[li];
        // FPS + gather centers
        k_fps<<<BB, FPS_T, 0, stream>>>(xyzL[li], N, S, fpsidx);
        k_gather_newxyz<<<(BB * S + 255) / 256, 256, 0, stream>>>(xyzL[li], fpsidx, N, S, nxyz[li]);

        int choff = 0;
        for (int br = 0; br < 2; ++br) {
            const int K = Karr[br];
            const float r2 = rad[li][br] * rad[li][br];
            const long M = (long)BB * S * K;          // multiple of 256

            k_ball_query<<<(BB * S) / 4, dim3(32, 4), 0, stream>>>(
                xyzL[li], nxyz[li], N, S, K, r2, gidx);

            const int CinL0 = C0[li] + 3;
            const int CinPad0 = pad32(CinL0);
            k_gather_group<<<(int)M, 64, 0, stream>>>(
                featsL[li], C0[li], fstride[li], xyzL[li], nxyz[li],
                gidx, N, S, K, CinPad0, bufA);

            _Float16* cur = bufA; _Float16* nxt = bufB;
            int cin = CinL0, cinPad = CinPad0;
            for (int ly = 0; ly < 3; ++ly) {
                const int cout = MLP[li][ly];
                const int coutPad = pad32(cout);
                {   // pack weights to padded f16
                    long tot = (long)coutPad * cinPad;
                    k_pack_w<<<(tot + 255) / 256, 256, 0, stream>>>(
                        P(li, br, ly, 0), cin, cout, cinPad, coutPad, Wp);
                }
                dim3 grid((unsigned)(M / 256), (unsigned)(coutPad / 32));
                k_gemm_bn_relu<<<grid, dim3(32, 4), 0, stream>>>(
                    cur, cinPad, Wp,
                    P(li, br, ly, 1), P(li, br, ly, 2),
                    P(li, br, ly, 3), P(li, br, ly, 4), P(li, br, ly, 5),
                    nxt, coutPad, cout);
                _Float16* t = cur; cur = nxt; nxt = t;
                cin = cout; cinPad = coutPad;
            }
            // after 3 layers the result is in `cur`
            k_maxpool<<<BB * S, 128, 0, stream>>>(
                cur, cinPad, cin, K, featsOut[li], CtotOut[li], choff);
            choff += cin;
        }
    }

    // outputs (return order): t(l1_xyz), t(l2_xyz), t(l3_xyz), t(l3_pts), t(l2_pts), t(l1_pts)
    float* out = (float*)d_out;
    long o = 0;
    {   long tot = (long)BB * 3 * 1024;
        k_transpose_out<<<(tot + 255) / 256, 256, 0, stream>>>(nxyz[0], out + o, 1024, 3); o += tot; }
    {   long tot = (long)BB * 3 * 512;
        k_transpose_out<<<(tot + 255) / 256, 256, 0, stream>>>(nxyz[1], out + o, 512, 3);  o += tot; }
    {   long tot = (long)BB * 3 * 256;
        k_transpose_out<<<(tot + 255) / 256, 256, 0, stream>>>(nxyz[2], out + o, 256, 3);  o += tot; }
    {   long tot = (long)BB * 512 * 256;
        k_transpose_out<<<(tot + 255) / 256, 256, 0, stream>>>(feats4, out + o, 256, 512); o += tot; }
    {   long tot = (long)BB * 256 * 512;
        k_transpose_out<<<(tot + 255) / 256, 256, 0, stream>>>(feats3, out + o, 512, 256); o += tot; }
    {   long tot = (long)BB * 64 * 1024;
        k_transpose_out<<<(tot + 255) / 256, 256, 0, stream>>>(feats2, out + o, 1024, 64); o += tot; }
}
